// MultiheadCosformerAttention__51642686767857
// MI455X (gfx1250) — matrix operand order repaired and checked by reading, not compiled, not run on hardware
//
#include <hip/hip_runtime.h>
#include <hip/hip_bf16.h>

// ---------------------------------------------------------------------------
// cosformer attention for MI455X (gfx1250), wave32 + WMMA bf16.
// All matmuls use v_wmma_f32_16x16x32_bf16 with fp32 accumulation.
// Bandwidth-bound op (~137 GFLOP, few hundred MB @ 23.3 TB/s), so:
//  - intermediates + weights converted to bf16 (halves HBM traffic),
//  - GLOBAL_LOAD_ASYNC_TO_LDS_B128 staging for pure-copy tiles (ASYNCcnt),
//  - b128 vectorized loads + LDS double-buffering in the big GEMM,
//  - sin/cos feature map precomputed once into a 32KB L2-resident table.
// ---------------------------------------------------------------------------

typedef __attribute__((ext_vector_type(16))) __bf16 v16bf;
typedef __attribute__((ext_vector_type(8)))  float  v8f;

#define PI_F 3.14159265358979323846f
#define ASYNC_STAGE 1

// Async global->LDS 16-byte copy (CDNA5 ISA §10.7 / §15.18.3, opcode 98).
// VDST = VGPR with wave-relative LDS byte address (= low 32 bits of the
// generic pointer, per ISA §10.2 aperture mapping), VADDR = 64-bit global VA.
__device__ __forceinline__ void async_copy_b128(const void* g, void* l) {
#if ASYNC_STAGE
  unsigned lds_addr = (unsigned)(unsigned long long)l;
  asm volatile("global_load_async_to_lds_b128 %0, %1, off"
               :: "v"(lds_addr), "v"(g) : "memory");
#else
  *(uint4*)l = *(const uint4*)g;
#endif
}
__device__ __forceinline__ void async_wait_all() {
#if ASYNC_STAGE
  asm volatile("s_wait_asynccnt 0" ::: "memory");
#endif
}

// --- WMMA fragment loaders (layouts per CDNA5 ISA 7.12.2, wave32) -----------
// A (16-bit, 16x32 MxK): lane half h = lane>>4; VGPRs 0..3 hold K in
// [8h, 8h+8), VGPRs 4..7 hold K in [16+8h, 16+8h+8). Row m = lane&15.
__device__ __forceinline__ v16bf frag_a(const __bf16* lds, int row, int ldk,
                                        int k0, int lane) {
  int h = (lane >> 4) & 1;
  const __bf16* p = lds + row * ldk + k0 + 8 * h;
  union { v16bf v; uint4 u[2]; } r;
  r.u[0] = *(const uint4*)p;          // K = k0+8h      .. +7
  r.u[1] = *(const uint4*)(p + 16);   // K = k0+16+8h   .. +7
  return r.v;
}
// B (16-bit, 32x16 KxN): lane col n = lane&15; same K order as A:
// elements 0..7 hold K in [8h, 8h+8), elements 8..15 hold K in [16+8h, 16+8h+8).
__device__ __forceinline__ v16bf frag_b(const __bf16* lds, int col, int ldk,
                                        int k0, int lane) {
  int h = (lane >> 4) & 1;
  const __bf16* p = lds + col * ldk + k0 + 8 * h;
  union { v16bf v; uint4 u[2]; } r;
  r.u[0] = *(const uint4*)p;          // K = k0+8h      .. +7
  r.u[1] = *(const uint4*)(p + 16);   // K = k0+16+8h   .. +7
  return r.v;
}

__device__ __forceinline__ v8f wmma_bf16(v16bf a, v16bf b, v8f c) {
  return __builtin_amdgcn_wmma_f32_16x16x32_bf16(
      false, a, false, b, (short)0, c, false, false);
}

// ---------------------------------------------------------------------------
// sin/cos feature-map table: sc[l] = {sin(ang_l), cos(ang_l)}. 32KB -> L2.
// ---------------------------------------------------------------------------
__global__ __launch_bounds__(256) void init_sincos(float2* __restrict__ sc,
                                                   int L) {
  int l = blockIdx.x * 256 + threadIdx.x;
  if (l < L) {
    float ang = PI_F * 0.5f * (float)(l + 1) / (float)L;
    sc[l] = make_float2(sinf(ang), cosf(ang));
  }
}

// fp32 -> bf16 one-time weight conversion (8MB bf16 total; negligible).
__global__ __launch_bounds__(256) void cvt_f32_bf16(const float* __restrict__ in,
                                                    void* __restrict__ out,
                                                    int n) {
  int g = blockIdx.x * 256 + threadIdx.x;
  if (g * 4 < n) {
    float4 v = *(const float4*)&in[g * 4];
    union { __bf16 h[4]; uint2 u; } pk;
    pk.h[0] = (__bf16)v.x; pk.h[1] = (__bf16)v.y;
    pk.h[2] = (__bf16)v.z; pk.h[3] = (__bf16)v.w;
    ((uint2*)out)[g] = pk.u;
  }
}

// ---------------------------------------------------------------------------
// Generic GEMM:  C[m,n] = act( sum_k A[m,k] * W[n,k] + bias[n] )
// Block tile 128x128, K-step 32, LDS double-buffered (1 barrier / K-step).
// W (bf16) staged via async global->LDS; A staged async when already bf16,
// else register-staged with fp32->bf16 convert. 8 waves in 4(M) x 2(N) grid;
// each wave owns 32x64 = 2x4 WMMA tiles.
// ---------------------------------------------------------------------------
template <bool RELU, bool A_BF16, bool OUT_BF16>
__global__ __launch_bounds__(256) void gemm_xWt(const void* __restrict__ Aptr,
                                                const void* __restrict__ Wptr,
                                                const float* __restrict__ bias,
                                                void* __restrict__ Cptr,
                                                int M, int N, int K) {
  __shared__ __align__(16) __bf16 As[2][128 * 40];
  __shared__ __align__(16) __bf16 Ws[2][128 * 40];
  const __bf16* W = (const __bf16*)Wptr;
  const int t = threadIdx.x, lane = t & 31, w = t >> 5;
  const int wm = w >> 1, wn = w & 1;  // 4x2 wave grid
  const int mb = blockIdx.x * 128, nb = blockIdx.y * 128;
  const int half = lane >> 4;

  v8f acc[2][4] = {};
  float4 areg_f[4];

  auto stage_async_W = [&](int k0, int buf) {  // 128x32 bf16, 2 b128/thread
#pragma unroll
    for (int i = 0; i < 2; ++i) {
      int idx = i * 256 + t;
      int row = idx >> 2, c8 = idx & 3;
      async_copy_b128(W + (size_t)(nb + row) * K + k0 + c8 * 8,
                      &Ws[buf][row * 40 + c8 * 8]);
    }
  };
  auto stage_async_A = [&](int k0, int buf) {  // bf16 A path
#pragma unroll
    for (int i = 0; i < 2; ++i) {
      int idx = i * 256 + t;
      int row = idx >> 2, c8 = idx & 3;
      async_copy_b128((const __bf16*)Aptr + (size_t)(mb + row) * K + k0 + c8 * 8,
                      &As[buf][row * 40 + c8 * 8]);
    }
  };
  auto load_regs_A = [&](int k0) {             // fp32 A path
#pragma unroll
    for (int i = 0; i < 4; ++i) {
      int idx = i * 256 + t;
      int row = idx >> 3, c4 = idx & 7;
      areg_f[i] = *(const float4*)((const float*)Aptr +
                                   (size_t)(mb + row) * K + k0 + c4 * 4);
    }
  };
  auto store_regs_A = [&](int buf) {
#pragma unroll
    for (int i = 0; i < 4; ++i) {
      int idx = i * 256 + t;
      int row = idx >> 3, c4 = idx & 7;
      union { __bf16 h[4]; uint2 u; } pk;
      pk.h[0] = (__bf16)areg_f[i].x; pk.h[1] = (__bf16)areg_f[i].y;
      pk.h[2] = (__bf16)areg_f[i].z; pk.h[3] = (__bf16)areg_f[i].w;
      *(uint2*)&As[buf][row * 40 + c4 * 4] = pk.u;
    }
  };

  // Prologue: stage tile k0=0 into buffer 0.
  stage_async_W(0, 0);
  if (A_BF16) {
    stage_async_A(0, 0);
  } else {
    load_regs_A(0);
    store_regs_A(0);
  }

  int cur = 0;
  for (int k0 = 0; k0 < K; k0 += 32) {
    async_wait_all();   // my async stores to LDS landed
    __syncthreads();    // everyone's landed; prev reads of cur^1 done
    const bool has_next = (k0 + 32) < K;
    if (has_next) {
      stage_async_W(k0 + 32, cur ^ 1);   // fire-and-forget, overlaps WMMAs
      if (A_BF16) stage_async_A(k0 + 32, cur ^ 1);
      else        load_regs_A(k0 + 32);
    }

    const __bf16* Ab = As[cur];
    const __bf16* Wb = Ws[cur];
    v16bf a0 = frag_a(Ab, wm * 32 + (lane & 15), 40, 0, lane);
    v16bf a1 = frag_a(Ab, wm * 32 + 16 + (lane & 15), 40, 0, lane);
    v16bf b0 = frag_b(Wb, wn * 64 + (lane & 15), 40, 0, lane);
    v16bf b1 = frag_b(Wb, wn * 64 + 16 + (lane & 15), 40, 0, lane);
    v16bf b2 = frag_b(Wb, wn * 64 + 32 + (lane & 15), 40, 0, lane);
    v16bf b3 = frag_b(Wb, wn * 64 + 48 + (lane & 15), 40, 0, lane);
    acc[0][0] = wmma_bf16(a0, b0, acc[0][0]);
    acc[0][1] = wmma_bf16(a0, b1, acc[0][1]);
    acc[0][2] = wmma_bf16(a0, b2, acc[0][2]);
    acc[0][3] = wmma_bf16(a0, b3, acc[0][3]);
    acc[1][0] = wmma_bf16(a1, b0, acc[1][0]);
    acc[1][1] = wmma_bf16(a1, b1, acc[1][1]);
    acc[1][2] = wmma_bf16(a1, b2, acc[1][2]);
    acc[1][3] = wmma_bf16(a1, b3, acc[1][3]);

    if (has_next && !A_BF16) store_regs_A(cur ^ 1);
    cur ^= 1;
  }

  for (int mt = 0; mt < 2; ++mt)
    for (int nt = 0; nt < 4; ++nt)
      for (int r = 0; r < 8; ++r) {
        int m = mb + wm * 32 + mt * 16 + r + 8 * half;
        int n = nb + wn * 64 + nt * 16 + (lane & 15);
        float v = acc[mt][nt][r] + bias[n];
        if (RELU) v = fmaxf(v, 0.0f);
        if (OUT_BF16)
          ((__bf16*)Cptr)[(size_t)m * N + n] = (__bf16)v;
        else
          ((float*)Cptr)[(size_t)m * N + n] = v;
      }
}

// ---------------------------------------------------------------------------
// phase_kv: grid (chunk 0..7, bh 0..63). Each block reduces 512 of L=4096:
//   kvpart[c,bh,d,m] = sum_{l in chunk} k_[l,d] * v[l,m]   (128x64)
//   kspart[c,bh,d]   = sum_{l in chunk} k_[l,d]
// k_[l,d] = k[l,d%64] * (d<64 ? sin_l : cos_l) via sc[] table.
// ---------------------------------------------------------------------------
__global__ __launch_bounds__(256) void phase_kv(const void* __restrict__ kbuf_,
                                                const void* __restrict__ vbuf_,
                                                const float2* __restrict__ sc,
                                                float* __restrict__ kvpart,
                                                float* __restrict__ kspart) {
  const __bf16* kbuf = (const __bf16*)kbuf_;
  const __bf16* vbuf = (const __bf16*)vbuf_;
  const int L = 4096, E = 1024, HD = 64, CH = 512;
  const int chunk = blockIdx.x;            // 0..7
  const int bh = blockIdx.y;               // 0..63
  const int b = bh >> 4, h = bh & 15;
  const int t = threadIdx.x, lane = t & 31, w = t >> 5;
  const int wm = w >> 1, wn = w & 1;
  const int half = lane >> 4;

  __shared__ __align__(16) __bf16 Ks[128 * 40];  // [d][l_local]
  __shared__ __align__(16) __bf16 Vs[64 * 40];   // [m][l_local]

  v8f acc[2][2] = {};
  float ksum_acc = 0.0f;
  const size_t base = ((size_t)b * L) * E + (size_t)h * HD;
  const int lbeg = chunk * CH;

  for (int l0 = lbeg; l0 < lbeg + CH; l0 += 32) {
    __syncthreads();
    {  // 256 threads stage 32l x 64d of k and v (one b128 load each)
      int dd8 = t & 7, ll = t >> 3;
      int l = l0 + ll;
      float2 s = sc[l];
      union { uint4 u; __bf16 hh[8]; } ku, vu;
      ku.u = *(const uint4*)(kbuf + base + (size_t)l * E + dd8 * 8);
      vu.u = *(const uint4*)(vbuf + base + (size_t)l * E + dd8 * 8);
#pragma unroll
      for (int j = 0; j < 8; ++j) {
        int dd = dd8 * 8 + j;
        float kval = (float)ku.hh[j];
        Ks[dd * 40 + ll]        = (__bf16)(kval * s.x);
        Ks[(64 + dd) * 40 + ll] = (__bf16)(kval * s.y);
        Vs[dd * 40 + ll]        = vu.hh[j];
      }
    }
    __syncthreads();
    if (t < 128) {
#pragma unroll 8
      for (int ll = 0; ll < 32; ++ll) ksum_acc += (float)Ks[t * 40 + ll];
    }
    v16bf a0 = frag_a(Ks, wm * 32 + (lane & 15), 40, 0, lane);
    v16bf a1 = frag_a(Ks, wm * 32 + 16 + (lane & 15), 40, 0, lane);
    v16bf b0 = frag_b(Vs, wn * 32 + (lane & 15), 40, 0, lane);
    v16bf b1 = frag_b(Vs, wn * 32 + 16 + (lane & 15), 40, 0, lane);
    acc[0][0] = wmma_bf16(a0, b0, acc[0][0]);
    acc[0][1] = wmma_bf16(a0, b1, acc[0][1]);
    acc[1][0] = wmma_bf16(a1, b0, acc[1][0]);
    acc[1][1] = wmma_bf16(a1, b1, acc[1][1]);
  }

  float* kvp = kvpart + ((size_t)(chunk * 64 + bh)) * (128 * 64);
  for (int mt = 0; mt < 2; ++mt)
    for (int nt = 0; nt < 2; ++nt)
      for (int r = 0; r < 8; ++r) {
        int d = wm * 32 + mt * 16 + r + 8 * half;
        int m = wn * 32 + nt * 16 + (lane & 15);
        kvp[(size_t)d * 64 + m] = acc[mt][nt][r];
      }
  if (t < 128) kspart[(size_t)(chunk * 64 + bh) * 128 + t] = ksum_acc;
}

// Deterministic sequential sum of the 8 chunk partials.
__global__ __launch_bounds__(256) void reduce_parts(
    const float* __restrict__ kvpart, const float* __restrict__ kspart,
    float* __restrict__ kv, float* __restrict__ ks) {
  const int NKV = 64 * 128 * 64;   // 524288
  const int NKS = 64 * 128;        // 8192
  int gid = blockIdx.x * 256 + threadIdx.x;
  if (gid < NKV) {
    float s = 0.0f;
#pragma unroll
    for (int c = 0; c < 8; ++c) s += kvpart[(size_t)c * NKV + gid];
    kv[gid] = s;
  } else if (gid < NKV + NKS) {
    int g2 = gid - NKV;
    float s = 0.0f;
#pragma unroll
    for (int c = 0; c < 8; ++c) s += kspart[(size_t)c * NKS + g2];
    ks[g2] = s;
  }
}

// ---------------------------------------------------------------------------
// phase_attn: block = (l-block of 128, bh). attn[l,m] = (q_ . kv[:,m]) * z[l],
// z[l] = 1/max(q_ . ksum, eps). Writes bf16 attn in (B,L,E) layout.
// ---------------------------------------------------------------------------
__global__ __launch_bounds__(256) void phase_attn(
    const void* __restrict__ qbuf_, const float* __restrict__ kv,
    const float* __restrict__ ksum, const float2* __restrict__ sc,
    void* __restrict__ obuf_) {
  const __bf16* qbuf = (const __bf16*)qbuf_;
  __bf16* obuf = (__bf16*)obuf_;
  const int L = 4096, E = 1024, HD = 64;
  const int l0 = blockIdx.x * 128;
  const int bh = blockIdx.y;
  const int b = bh >> 4, h = bh & 15;
  const int t = threadIdx.x, lane = t & 31, w = t >> 5;
  const int wm = w >> 1, wn = w & 1;
  const int half = lane >> 4;

  __shared__ __align__(16) __bf16 Qs[128 * 136];   // [l_local][d]
  __shared__ __align__(16) __bf16 KVs[64 * 136];   // [m][d]
  __shared__ float ksums[128];
  __shared__ float zs[128];

  const size_t base = ((size_t)b * L) * E + (size_t)h * HD;

#pragma unroll
  for (int i = 0; i < 4; ++i) {
    int idx = i * 256 + t;           // 0..1023 groups of 8
    int dd8 = idx & 7, ll = idx >> 3;
    int l = l0 + ll;
    float2 s = sc[l];
    union { uint4 u; __bf16 hh[8]; } qu, so, co;
    qu.u = *(const uint4*)(qbuf + base + (size_t)l * E + dd8 * 8);
#pragma unroll
    for (int j = 0; j < 8; ++j) {
      float qv = (float)qu.hh[j];
      so.hh[j] = (__bf16)(qv * s.x);
      co.hh[j] = (__bf16)(qv * s.y);
    }
    *(uint4*)&Qs[ll * 136 + dd8 * 8]      = so.u;
    *(uint4*)&Qs[ll * 136 + 64 + dd8 * 8] = co.u;
  }
#pragma unroll
  for (int i = 0; i < 8; ++i) {
    int idx = i * 256 + t;           // 0..2047 float4 groups
    int m4 = idx & 15, d = idx >> 4;
    float4 vv = *(const float4*)&kv[(size_t)bh * 8192 + (size_t)d * 64 + m4 * 4];
    KVs[(m4 * 4 + 0) * 136 + d] = (__bf16)vv.x;
    KVs[(m4 * 4 + 1) * 136 + d] = (__bf16)vv.y;
    KVs[(m4 * 4 + 2) * 136 + d] = (__bf16)vv.z;
    KVs[(m4 * 4 + 3) * 136 + d] = (__bf16)vv.w;
  }
  if (t < 128) ksums[t] = ksum[bh * 128 + t];
  __syncthreads();

  if (t < 128) {
    float s = 0.0f;
#pragma unroll 8
    for (int d = 0; d < 128; ++d) s += (float)Qs[t * 136 + d] * ksums[d];
    zs[t] = 1.0f / fmaxf(s, 1e-4f);
  }
  __syncthreads();

  v8f acc[2][2] = {};
#pragma unroll
  for (int ks2 = 0; ks2 < 128; ks2 += 32) {
    v16bf a0 = frag_a(Qs, wm * 32 + (lane & 15), 136, ks2, lane);
    v16bf a1 = frag_a(Qs, wm * 32 + 16 + (lane & 15), 136, ks2, lane);
    v16bf b0 = frag_b(KVs, wn * 32 + (lane & 15), 136, ks2, lane);
    v16bf b1 = frag_b(KVs, wn * 32 + 16 + (lane & 15), 136, ks2, lane);
    acc[0][0] = wmma_bf16(a0, b0, acc[0][0]);
    acc[0][1] = wmma_bf16(a0, b1, acc[0][1]);
    acc[1][0] = wmma_bf16(a1, b0, acc[1][0]);
    acc[1][1] = wmma_bf16(a1, b1, acc[1][1]);
  }

  for (int mt = 0; mt < 2; ++mt)
    for (int nt = 0; nt < 2; ++nt)
      for (int r = 0; r < 8; ++r) {
        int ml = wm * 32 + mt * 16 + r + 8 * half;
        int m  = wn * 32 + nt * 16 + (lane & 15);
        float v = acc[mt][nt][r] * zs[ml];
        obuf[base + (size_t)(l0 + ml) * E + m] = (__bf16)v;
      }
}

// ---------------------------------------------------------------------------
extern "C" void kernel_launch(void* const* d_in, const int* in_sizes, int n_in,
                              void* d_out, int out_size, void* d_ws,
                              size_t ws_size, hipStream_t stream) {
  const float* x  = (const float*)d_in[0];
  const float* Wq = (const float*)d_in[1];
  const float* bq = (const float*)d_in[2];
  const float* Wk = (const float*)d_in[3];
  const float* bk = (const float*)d_in[4];
  const float* Wv = (const float*)d_in[5];
  const float* bv = (const float*)d_in[6];
  const float* Wo = (const float*)d_in[7];
  const float* bo = (const float*)d_in[8];

  const int B = 4, L = 4096, E = 1024, H = 16;
  const int M = B * L;  // 16384

  // Workspace layout.
  char* ws = (char*)d_ws;
  size_t off = 0;
  void* qb = ws + off; off += (size_t)M * E * 2;            // 32MB bf16
  void* kb = ws + off; off += (size_t)M * E * 2;            // 32MB
  void* vb = ws + off; off += (size_t)M * E * 2;            // 32MB
  void* ob = ws + off; off += (size_t)M * E * 2;            // 32MB
  float* kv     = (float*)(ws + off); off += (size_t)64 * 128 * 64 * 4;     // 2MB
  float* ksum   = (float*)(ws + off); off += (size_t)64 * 128 * 4;          // 32KB
  float* kvpart = (float*)(ws + off); off += (size_t)8 * 64 * 128 * 64 * 4; // 16MB
  float* kspart = (float*)(ws + off); off += (size_t)8 * 64 * 128 * 4;      // 256KB
  float2* sc    = (float2*)(ws + off); off += (size_t)L * 8;                // 32KB
  void* wq_bf = ws + off; off += (size_t)E * E * 2;         // 2MB bf16 weights
  void* wk_bf = ws + off; off += (size_t)E * E * 2;
  void* wv_bf = ws + off; off += (size_t)E * E * 2;
  void* wo_bf = ws + off; off += (size_t)E * E * 2;

  dim3 blk(256);
  dim3 gg(M / 128, E / 128);     // 128 x 8 blocks for the big GEMMs
  dim3 gcvt(E * E / 1024);       // 1M elems, 4/thread

  init_sincos<<<dim3(L / 256), blk, 0, stream>>>(sc, L);
  cvt_f32_bf16<<<gcvt, blk, 0, stream>>>(Wq, wq_bf, E * E);
  cvt_f32_bf16<<<gcvt, blk, 0, stream>>>(Wk, wk_bf, E * E);
  cvt_f32_bf16<<<gcvt, blk, 0, stream>>>(Wv, wv_bf, E * E);
  cvt_f32_bf16<<<gcvt, blk, 0, stream>>>(Wo, wo_bf, E * E);

  gemm_xWt<true,  false, true><<<gg, blk, 0, stream>>>(x, wq_bf, bq, qb, M, E, E);
  gemm_xWt<true,  false, true><<<gg, blk, 0, stream>>>(x, wk_bf, bk, kb, M, E, E);
  gemm_xWt<false, false, true><<<gg, blk, 0, stream>>>(x, wv_bf, bv, vb, M, E, E);

  phase_kv<<<dim3(8, B * H), blk, 0, stream>>>(kb, vb, sc, kvpart, kspart);
  reduce_parts<<<dim3((64 * 128 * 64 + 64 * 128) / 256), blk, 0, stream>>>(
      kvpart, kspart, kv, ksum);
  phase_attn<<<dim3(L / 128, B * H), blk, 0, stream>>>(qb, kv, ksum, sc, ob);

  gemm_xWt<false, true, false><<<gg, blk, 0, stream>>>(ob, wo_bf, bo, d_out,
                                                       M, E, E);
}
